// SegmentLinear_72189810312114
// MI455X (gfx1250) — 8-bit matrix operand order repaired and checked by reading, compiled, not run on hardware
//
#include <hip/hip_runtime.h>

typedef __attribute__((ext_vector_type(8))) int v8i;

#define M_DIM 8192
#define N_DIM 6144
#define K_DIM 2048
#define CHUNK 2048

#define BM 128
#define BN 128
#define BK 64
#define KSTEPS (K_DIM / BK)
#define LDS_P 80

__global__ void zero_amax_kernel(unsigned int* amax) {
    if (threadIdx.x < 4) amax[threadIdx.x] = 0u;
}

__device__ __forceinline__ float abs4max(float4 v) {
    return fmaxf(fmaxf(fabsf(v.x), fabsf(v.y)), fmaxf(fabsf(v.z), fabsf(v.w)));
}

__device__ __forceinline__ void block_amax_commit(float m, unsigned int* slot) {
    __shared__ float red[256];
    red[threadIdx.x] = m;
    __syncthreads();
    for (int s = 128; s > 0; s >>= 1) {
        if ((int)threadIdx.x < s) red[threadIdx.x] = fmaxf(red[threadIdx.x], red[threadIdx.x + s]);
        __syncthreads();
    }
    if (threadIdx.x == 0) atomicMax(slot, __float_as_uint(red[0]));
}

__global__ __launch_bounds__(256) void amax_x_kernel(const float4* __restrict__ x,
                                                     unsigned int* __restrict__ amax) {
    size_t t = (size_t)blockIdx.x * 256 + threadIdx.x;
    const float4* p = x + t * 4;
    float m = 0.0f;
#pragma unroll
    for (int i = 0; i < 4; ++i) m = fmaxf(m, abs4max(p[i]));
    block_amax_commit(m, &amax[0]);
}

__global__ __launch_bounds__(256) void amax_w_kernel(const float4* __restrict__ w,
                                                     unsigned int* __restrict__ amax) {
    size_t t = (size_t)blockIdx.x * 256 + threadIdx.x;
    const float4* p = w + t * 4;
    float m = 0.0f;
#pragma unroll
    for (int i = 0; i < 4; ++i) m = fmaxf(m, abs4max(p[i]));
    block_amax_commit(m, &amax[1 + blockIdx.x / 1024]);
}

__device__ __forceinline__ int q8(float v, float inv) {
    int q = __float2int_rn(v * inv);
    return min(127, max(-127, q));
}

__device__ __forceinline__ int pack4(float4 v, float inv) {
    int a = q8(v.x, inv), b = q8(v.y, inv), c = q8(v.z, inv), d = q8(v.w, inv);
    return (a & 255) | ((b & 255) << 8) | ((c & 255) << 16) | ((d & 255) << 24);
}

__global__ __launch_bounds__(256) void quant_x_kernel(const float4* __restrict__ x,
                                                      const unsigned int* __restrict__ amax,
                                                      int4* __restrict__ xq) {
    float scale = __uint_as_float(amax[0]) * (1.0f / 127.0f);
    float inv = scale > 0.0f ? 1.0f / scale : 0.0f;
    size_t t = (size_t)blockIdx.x * 256 + threadIdx.x;
    const float4* p = x + t * 4;
    int4 o;
    o.x = pack4(p[0], inv); o.y = pack4(p[1], inv);
    o.z = pack4(p[2], inv); o.w = pack4(p[3], inv);
    xq[t] = o;
}

__global__ __launch_bounds__(256) void quant_w_kernel(const float4* __restrict__ w,
                                                      const unsigned int* __restrict__ amax,
                                                      int4* __restrict__ wq) {
    float scale = __uint_as_float(amax[1 + blockIdx.x / 1024]) * (1.0f / 127.0f);
    float inv = scale > 0.0f ? 1.0f / scale : 0.0f;
    size_t t = (size_t)blockIdx.x * 256 + threadIdx.x;
    const float4* p = w + t * 4;
    int4 o;
    o.x = pack4(p[0], inv); o.y = pack4(p[1], inv);
    o.z = pack4(p[2], inv); o.w = pack4(p[3], inv);
    wq[t] = o;
}

__device__ __forceinline__ void async_cp16(unsigned lds_off, const signed char* g) {
    unsigned long long ga = (unsigned long long)g;
    asm volatile("global_load_async_to_lds_b128 %0, %1, off"
                 :: "v"(lds_off), "v"(ga) : "memory");
}

__device__ __forceinline__ void wait_async0() {
    asm volatile("s_wait_asynccnt 0x0" ::: "memory");
}

__global__ __launch_bounds__(256) void gemm_i8_kernel(const signed char* __restrict__ xq,
                                                      const signed char* __restrict__ wq,
                                                      const float* __restrict__ bias,
                                                      const unsigned int* __restrict__ amax,
                                                      float* __restrict__ out) {
    __shared__ __align__(16) char lds_a[2][BM * LDS_P];
    __shared__ __align__(16) char lds_b[2][BN * LDS_P];

    const int tid    = threadIdx.x;
    const int lane   = tid & 31;
    const int wid    = tid >> 5;
    const int wave_m = wid & 3;
    const int wave_n = wid >> 2;
    const int half   = lane >> 4;
    const int l16    = lane & 15;

    const int block_m = blockIdx.y * BM;
    const int block_n = blockIdx.x * BN;

    const float sx    = __uint_as_float(amax[0]) * (1.0f / 127.0f);
    const float sw    = __uint_as_float(amax[1 + block_n / CHUNK]) * (1.0f / 127.0f);
    const float scale = sx * sw;

    v8i acc[2][4];
#pragma unroll
    for (int sm = 0; sm < 2; ++sm)
#pragma unroll
        for (int sn = 0; sn < 4; ++sn)
#pragma unroll
            for (int r = 0; r < 8; ++r) acc[sm][sn][r] = 0;

    const int sr = tid >> 2;
    const int sc = (tid & 3) * 16;

#pragma unroll
    for (int p = 0; p < 2; ++p) {
        const int r = sr + p * 64;
        async_cp16((unsigned)(size_t)&lds_a[0][r * LDS_P + sc],
                   xq + (size_t)(block_m + r) * K_DIM + sc);
        async_cp16((unsigned)(size_t)&lds_b[0][r * LDS_P + sc],
                   wq + (size_t)(block_n + r) * K_DIM + sc);
    }
    wait_async0();
    __syncthreads();

    for (int kb = 0; kb < KSTEPS; ++kb) {
        const int cur = kb & 1;
        if (kb + 1 < KSTEPS) {
            const int k1 = (kb + 1) * BK;
#pragma unroll
            for (int p = 0; p < 2; ++p) {
                const int r = sr + p * 64;
                async_cp16((unsigned)(size_t)&lds_a[cur ^ 1][r * LDS_P + sc],
                           xq + (size_t)(block_m + r) * K_DIM + k1 + sc);
                async_cp16((unsigned)(size_t)&lds_b[cur ^ 1][r * LDS_P + sc],
                           wq + (size_t)(block_n + r) * K_DIM + k1 + sc);
            }
        }

        v8i afrag[2];
#pragma unroll
        for (int sm = 0; sm < 2; ++sm) {
            const char* rp = &lds_a[cur][(wave_m * 32 + sm * 16 + l16) * LDS_P] + half * 16;
            unsigned long long q0 = *(const unsigned long long*)(rp + 0);
            unsigned long long q1 = *(const unsigned long long*)(rp + 8);
            unsigned long long q2 = *(const unsigned long long*)(rp + 32);
            unsigned long long q3 = *(const unsigned long long*)(rp + 40);
            v8i a;
            a[0] = (int)q0; a[1] = (int)(q0 >> 32);
            a[2] = (int)q1; a[3] = (int)(q1 >> 32);
            a[4] = (int)q2; a[5] = (int)(q2 >> 32);
            a[6] = (int)q3; a[7] = (int)(q3 >> 32);
            afrag[sm] = a;
        }

        v8i bfrag[4];
#pragma unroll
        for (int sn = 0; sn < 4; ++sn) {
            const char* rp = &lds_b[cur][(wave_n * 64 + sn * 16 + l16) * LDS_P] + half * 16;
            int4 p0 = *(const int4*)(rp + 0);
            int4 p1 = *(const int4*)(rp + 32);
            v8i b;
            b[0] = p0.x; b[1] = p0.y; b[2] = p0.z; b[3] = p0.w;
            b[4] = p1.x; b[5] = p1.y; b[6] = p1.z; b[7] = p1.w;
            bfrag[sn] = b;
        }

#pragma unroll
        for (int sn = 0; sn < 4; ++sn)
#pragma unroll
            for (int sm = 0; sm < 2; ++sm)
                acc[sm][sn] = __builtin_amdgcn_wmma_i32_16x16x64_iu8(
                    true, afrag[sm], true, bfrag[sn], acc[sm][sn], false, false);

        wait_async0();
        __syncthreads();
    }

#pragma unroll
    for (int sn = 0; sn < 4; ++sn) {
        const int n  = block_n + wave_n * 64 + sn * 16 + l16;
        const float bv = bias[n];
#pragma unroll
        for (int sm = 0; sm < 2; ++sm) {
            const int mbase = block_m + wave_m * 32 + sm * 16 + half * 8;
            v8i c = acc[sm][sn];
#pragma unroll
            for (int r = 0; r < 8; ++r) {
                out[(size_t)(mbase + r) * N_DIM + n] = (float)c[r] * scale + bv;
            }
        }
    }
}

extern "C" void kernel_launch(void* const* d_in, const int* in_sizes, int n_in,
                              void* d_out, int out_size, void* d_ws, size_t ws_size,
                              hipStream_t stream) {
    const float* x = (const float*)d_in[0];
    const float* W = (const float*)d_in[1];
    const float* b = (const float*)d_in[2];
    float* out = (float*)d_out;

    unsigned int* amax = (unsigned int*)d_ws;
    signed char* xq = (signed char*)d_ws + 256;
    signed char* wq = xq + (size_t)M_DIM * K_DIM;

    zero_amax_kernel<<<1, 32, 0, stream>>>(amax);
    amax_x_kernel<<<4096, 256, 0, stream>>>((const float4*)x, amax);
    amax_w_kernel<<<3072, 256, 0, stream>>>((const float4*)W, amax);
    quant_x_kernel<<<4096, 256, 0, stream>>>((const float4*)x, amax, (int4*)xq);
    quant_w_kernel<<<3072, 256, 0, stream>>>((const float4*)W, amax, (int4*)wq);

    dim3 grid(N_DIM / BN, M_DIM / BM);
    gemm_i8_kernel<<<grid, 256, 0, stream>>>(xq, wq, b, amax, out);
}
